// TransformerInFeat_41369124995488
// MI455X (gfx1250) — matrix operand order repaired and checked by reading, not compiled, not run on hardware
//
#include <hip/hip_runtime.h>
#include <math.h>

typedef __attribute__((ext_vector_type(16))) __bf16 v16bf;
typedef __attribute__((ext_vector_type(8)))  __bf16 v8bf;
typedef __attribute__((ext_vector_type(8)))  float  v8f;

union AOp {
    v16bf v;
    v8bf  h[2];
    unsigned int u[8];
};

// ---------------------------------------------------------------------------
// zero a small float region
__global__ void zero_kernel(float* __restrict__ p, int n) {
    int t = blockIdx.x * blockDim.x + threadIdx.x;
    if (t < n) p[t] = 0.0f;
}

// ---------------------------------------------------------------------------
// segment-sum pooling: pooled[b][c] += feats[r][c], counts[b] += 1
__global__ void pool_kernel(const float* __restrict__ feats, const int* __restrict__ bidx,
                            int N, float* __restrict__ pooled, float* __restrict__ counts) {
    __shared__ float sm[256];
    __shared__ float cs[16];
    const int c   = threadIdx.x & 63;
    const int sub = threadIdx.x >> 6;   // 0..3
    float a[4] = {0.f, 0.f, 0.f, 0.f};
    float n[4] = {0.f, 0.f, 0.f, 0.f};
    for (int r = blockIdx.x * 4 + sub; r < N; r += gridDim.x * 4) {
        const int   b = bidx[r];
        const float v = feats[((size_t)r << 6) + c];
        if      (b == 0) { a[0] += v; n[0] += 1.f; }
        else if (b == 1) { a[1] += v; n[1] += 1.f; }
        else if (b == 2) { a[2] += v; n[2] += 1.f; }
        else             { a[3] += v; n[3] += 1.f; }
    }
    for (int b = 0; b < 4; ++b) {
        sm[threadIdx.x] = a[b];
        __syncthreads();
        if (sub == 0)
            atomicAdd(&pooled[b * 64 + c], sm[c] + sm[64 + c] + sm[128 + c] + sm[192 + c]);
        __syncthreads();
    }
    if (c == 0) {
        for (int b = 0; b < 4; ++b) cs[sub * 4 + b] = n[b];
    }
    __syncthreads();
    if (threadIdx.x < 4) {
        const int b = threadIdx.x;
        atomicAdd(&counts[b], cs[b] + cs[4 + b] + cs[8 + b] + cs[12 + b]);
    }
}

// ---------------------------------------------------------------------------
// SE gate: mean -> fc1 -> relu -> fc2 -> sigmoid  (B=4, C=64, R=16; 1 block of 64)
__global__ void gate_kernel(const float* __restrict__ pooled, const float* __restrict__ counts,
                            const float* __restrict__ w1, const float* __restrict__ bf1,
                            const float* __restrict__ w2, const float* __restrict__ bf2,
                            float* __restrict__ gate) {
    __shared__ float pm[4 * 64];
    __shared__ float h[4 * 16];
    const int t = threadIdx.x;  // 0..63
    for (int b = 0; b < 4; ++b) pm[b * 64 + t] = pooled[b * 64 + t] / counts[b];
    __syncthreads();
    {
        const int b = t >> 4, r = t & 15;
        float acc = bf1[r];
        for (int cc = 0; cc < 64; ++cc) acc += pm[b * 64 + cc] * w1[cc * 16 + r];
        h[t] = acc > 0.f ? acc : 0.f;
    }
    __syncthreads();
    for (int bb = 0; bb < 4; ++bb) {
        float a = bf2[t];
        for (int rr = 0; rr < 16; ++rr) a += h[bb * 16 + rr] * w2[rr * 64 + t];
        gate[bb * 64 + t] = 1.0f / (1.0f + expf(-a));
    }
}

// ---------------------------------------------------------------------------
// y_bf16[r][c] = bf16(feats[r][c] * gate[batch[r]][c])  (4 elems / thread)
__global__ void scale_kernel(const float* __restrict__ feats, const int* __restrict__ bidx,
                             const float* __restrict__ gate, __bf16* __restrict__ y,
                             long total4) {
    const long stride = (long)gridDim.x * blockDim.x;
    for (long i = blockIdx.x * (long)blockDim.x + threadIdx.x; i < total4; i += stride) {
        const long e = i << 2;
        const int  r = (int)(e >> 6);
        const int  c = (int)(e & 63);
        const int  b = bidx[r];
        const float4 f = *(const float4*)(feats + e);
        const float* g = gate + (b << 6) + c;
        union { __bf16 b4[4]; unsigned long long u; } o;
        o.b4[0] = (__bf16)(f.x * g[0]);
        o.b4[1] = (__bf16)(f.y * g[1]);
        o.b4[2] = (__bf16)(f.z * g[2]);
        o.b4[3] = (__bf16)(f.w * g[3]);
        *(unsigned long long*)(y + e) = o.u;
    }
}

// ---------------------------------------------------------------------------
// Pack W [27][Cin=64][Cout=64] f32 -> B-operand layout [k][kc][nt][lane][16] bf16.
// Element j of lane l (v16bf): K = kc*32 + (l>=16 ? 8 : 0) + (j&7) + (j>=8 ? 16 : 0), N = nt*16 + (l&15).
__global__ void pack_w_kernel(const float* __restrict__ w, __bf16* __restrict__ wp) {
    const int t = blockIdx.x * blockDim.x + threadIdx.x;
    if (t >= 27 * 2 * 4 * 32 * 16) return;
    const int j    = t & 15;
    const int lane = (t >> 4) & 31;
    const int nt   = (t >> 9) & 3;
    const int kc   = (t >> 11) & 1;
    const int k    = t >> 12;
    const int cin  = kc * 32 + ((lane >= 16) ? 8 : 0) + (j & 7) + ((j >= 8) ? 16 : 0);
    const int cout = nt * 16 + (lane & 15);
    wp[t] = (__bf16)w[((size_t)k * 64 + cin) * 64 + cout];
}

// ---------------------------------------------------------------------------
// Sparse 3x3x3 conv via WMMA bf16. One wave = 32 output rows x 64 output cols
// (two 16-row M-tiles sharing every B operand -> 16 WMMA per 18 VMEM per k).
// MODE 0: BN(gamma,beta,mean,var) + ReLU -> bf16 out. MODE 1: +bias -> f32 out.
template <int MODE>
__global__ __launch_bounds__(128, 1) void spconv_wmma_kernel(
    const __bf16* __restrict__ x,    // [N,64] bf16 input features
    const __bf16* __restrict__ wp,   // packed weights [27][2][4][32][16]
    const int*    __restrict__ nbr,  // [27][N]
    int N,
    const float* __restrict__ e0, const float* __restrict__ e1,
    const float* __restrict__ e2, const float* __restrict__ e3,
    __bf16* __restrict__ out_bf, float* __restrict__ out_f) {
    const int lane = threadIdx.x & 31;
    const int wave = threadIdx.x >> 5;
    const int tile = blockIdx.x * 4 + wave;   // 32-row tile
    const int base = tile << 5;
    if (base >= N) return;  // wave-uniform exit

    const int rc  = lane & 15;   // A: row-in-subtile; B/D: col-in-ntile
    const int hi  = lane >> 4;
    const int seg = hi << 3;     // A K-segment offset / D row offset
    const int gr0 = base + rc;
    const int gr1 = base + 16 + rc;

    v8f acc[2][4] = {};

    for (int k = 0; k < 27; ++k) {
        const int* nk = nbr + (size_t)k * N;
        const int id0 = (gr0 < N) ? nk[gr0] : -1;
        const int id1 = (gr1 < N) ? nk[gr1] : -1;
        const __bf16* __restrict__ row0 = x + ((size_t)(id0 < 0 ? 0 : id0) << 6);
        const __bf16* __restrict__ row1 = x + ((size_t)(id1 < 0 ? 0 : id1) << 6);
        const __bf16* __restrict__ wk   = wp + ((size_t)k << 12);
#pragma unroll
        for (int kc = 0; kc < 2; ++kc) {
            AOp a0, a1;
            if (id0 >= 0) {
                // lane<16: K {0..7, 16..23}; lane>=16: K {8..15, 24..31} of this chunk
                a0.h[0] = *(const v8bf*)(row0 + kc * 32 + seg);
                a0.h[1] = *(const v8bf*)(row0 + kc * 32 + seg + 16);
            } else {
#pragma unroll
                for (int q = 0; q < 8; ++q) a0.u[q] = 0u;
            }
            if (id1 >= 0) {
                a1.h[0] = *(const v8bf*)(row1 + kc * 32 + seg);
                a1.h[1] = *(const v8bf*)(row1 + kc * 32 + seg + 16);
            } else {
#pragma unroll
                for (int q = 0; q < 8; ++q) a1.u[q] = 0u;
            }
            const __bf16* __restrict__ wkc = wk + (kc << 11) + (lane << 4);
#pragma unroll
            for (int nt = 0; nt < 4; ++nt) {
                const v16bf b = *(const v16bf*)(wkc + (nt << 9));
                acc[0][nt] = __builtin_amdgcn_wmma_f32_16x16x32_bf16(
                    false, a0.v, false, b, (short)0, acc[0][nt], false, false);
                acc[1][nt] = __builtin_amdgcn_wmma_f32_16x16x32_bf16(
                    false, a1.v, false, b, (short)0, acc[1][nt], false, false);
            }
        }
    }

    // Epilogue: lane holds column col = nt*16 + rc; rows m = base + t*16 + seg + r.
#pragma unroll
    for (int t = 0; t < 2; ++t) {
#pragma unroll
        for (int nt = 0; nt < 4; ++nt) {
            const int col = (nt << 4) + rc;
            if (MODE == 0) {
                const float sc = e0[col] * rsqrtf(e3[col] + 1e-5f);
                const float sh = e1[col] - e2[col] * sc;
#pragma unroll
                for (int r = 0; r < 8; ++r) {
                    const int m = base + (t << 4) + seg + r;
                    if (m < N) {
                        const float v = fmaf(acc[t][nt][r], sc, sh);
                        out_bf[((size_t)m << 6) + col] = (__bf16)(v > 0.f ? v : 0.f);
                    }
                }
            } else {
                const float bias = e0[col];
#pragma unroll
                for (int r = 0; r < 8; ++r) {
                    const int m = base + (t << 4) + seg + r;
                    if (m < N) out_f[((size_t)m << 6) + col] = acc[t][nt][r] + bias;
                }
            }
        }
    }
}

// ---------------------------------------------------------------------------
extern "C" void kernel_launch(void* const* d_in, const int* in_sizes, int n_in,
                              void* d_out, int out_size, void* d_ws, size_t ws_size,
                              hipStream_t stream) {
    const float* feats    = (const float*)d_in[0];
    const int*   nbr      = (const int*)d_in[1];
    const int*   bidx     = (const int*)d_in[2];
    const float* w_fc1    = (const float*)d_in[3];
    const float* b_fc1    = (const float*)d_in[4];
    const float* w_fc2    = (const float*)d_in[5];
    const float* b_fc2    = (const float*)d_in[6];
    const float* w_conv1  = (const float*)d_in[7];
    const float* bn_gamma = (const float*)d_in[8];
    const float* bn_beta  = (const float*)d_in[9];
    const float* bn_mean  = (const float*)d_in[10];
    const float* bn_var   = (const float*)d_in[11];
    const float* w_conv2  = (const float*)d_in[12];
    const float* b_conv2  = (const float*)d_in[13];

    const int N = in_sizes[0] / 64;

    char*   ws     = (char*)d_ws;
    float*  pooled = (float*)(ws);                       // 256 f32
    float*  counts = (float*)(ws + 1024);                // 4 f32
    float*  gate   = (float*)(ws + 2048);                // 256 f32
    __bf16* wp1    = (__bf16*)(ws + 4096);               // 221184 B
    __bf16* wp2    = (__bf16*)(ws + 4096 + 262144);      // 221184 B
    __bf16* ybf    = (__bf16*)(ws + (1 << 20));          // N*64*2 B
    __bf16* hbf    = (__bf16*)(ws + (1 << 20) + (size_t)N * 128);

    // 1) zero pooled+counts
    zero_kernel<<<1, 512, 0, stream>>>(pooled, 512);
    // 2) segment pooling
    pool_kernel<<<512, 256, 0, stream>>>(feats, bidx, N, pooled, counts);
    // 3) SE gate
    gate_kernel<<<1, 64, 0, stream>>>(pooled, counts, w_fc1, b_fc1, w_fc2, b_fc2, gate);
    // 4) gated features -> bf16
    {
        const long total4 = (long)N * 16;
        scale_kernel<<<2048, 256, 0, stream>>>(feats, bidx, gate, ybf, total4);
    }
    // 5) pack weights into WMMA B-operand layout
    pack_w_kernel<<<432, 256, 0, stream>>>(w_conv1, wp1);
    pack_w_kernel<<<432, 256, 0, stream>>>(w_conv2, wp2);
    // 6) conv1 + BN + ReLU -> bf16
    const int tiles  = (N + 31) / 32;
    const int blocks = (tiles + 3) / 4;
    spconv_wmma_kernel<0><<<blocks, 128, 0, stream>>>(
        ybf, wp1, nbr, N, bn_gamma, bn_beta, bn_mean, bn_var, hbf, nullptr);
    // 7) conv2 + bias -> f32 output
    spconv_wmma_kernel<1><<<blocks, 128, 0, stream>>>(
        hbf, wp2, nbr, N, b_conv2, nullptr, nullptr, nullptr, nullptr, (float*)d_out);
}
